// MultiHeadAttention_43396349558876
// MI455X (gfx1250) — matrix operand order repaired and checked by reading, not compiled, not run on hardware
//
#include <hip/hip_runtime.h>
#include <hip/hip_bf16.h>

// MHA forward for MI455X (gfx1250, wave32, WMMA + async LDS copies).
// Sizes fixed by the reference: B=4, S=2048, E=1024, H=16, D=64.

typedef __attribute__((ext_vector_type(16))) _Float16 v16h;
typedef __attribute__((ext_vector_type(8)))  _Float16 v8h;
typedef __attribute__((ext_vector_type(4)))  _Float16 v4h;
typedef __attribute__((ext_vector_type(8)))  float    v8f;

#define Bn 4
#define Sn 2048
#define En 1024
#define Hn 16
#define Dn 64
#define Mn (Bn * Sn)   // 8192 rows for the projection GEMMs

__device__ __forceinline__ v8f wmma_f16(v16h a, v16h b, v8f c) {
  // D = A(16x32 f16) * B(32x16 f16) + C(16x16 f32)
  return __builtin_amdgcn_wmma_f32_16x16x32_f16(
      /*neg_a=*/false, a, /*neg_b=*/false, b,
      /*c_mod=*/(short)0, c, /*reuse_a=*/false, /*reuse_b=*/false);
}

// A-operand fragment: lane (L&15) holds row L&15; K-halves {0..15} for
// lanes 0-15 and {16..31} for lanes 16-31 -> two contiguous 16B chunks.
__device__ __forceinline__ v16h load_a16(const _Float16* lo, const _Float16* hi) {
  v16h r;
  ((v8h*)&r)[0] = *(const v8h*)lo;
  ((v8h*)&r)[1] = *(const v8h*)hi;
  return r;
}

// Generic pointer to a __shared__ object: low 32 bits are the LDS byte offset
// (ISA 10.2: LDS aperture address maps via addr[31:0]).
__device__ __forceinline__ uint32_t lds_off(const void* p) {
  return (uint32_t)(uintptr_t)p;
}

// One wave-instruction async copy: each lane moves 16B global -> LDS.
// Tracked by ASYNCcnt; no VGPR data, overlaps with WMMA issue.
__device__ __forceinline__ void async_copy_b128(uint32_t lds, const _Float16* g) {
  asm volatile("global_load_async_to_lds_b128 %0, %1, off"
               :: "v"(lds), "v"(g)
               : "memory");
}

__device__ __forceinline__ void wait_async0() {
  asm volatile("s_wait_asynccnt 0x0" ::: "memory");
}

// ---------------------------------------------------------------------------
// fp32 -> f16 conversion (vectorized x4)
// ---------------------------------------------------------------------------
__global__ __launch_bounds__(256) void cvt4_kernel(const float* __restrict__ in,
                                                   _Float16* __restrict__ out,
                                                   int n4) {
  int i = blockIdx.x * 256 + threadIdx.x;
  if (i >= n4) return;
  float4 v = ((const float4*)in)[i];
  v4h h = {(_Float16)v.x, (_Float16)v.y, (_Float16)v.z, (_Float16)v.w};
  ((v4h*)out)[i] = h;
}

// ---------------------------------------------------------------------------
// GEMM: Y[M,N] = A[M,K] @ W[N,K]^T + bias   (A,W f16 row-major, acc fp32)
//   mode 0: store f16, Q/K layout [b,h,s,d]
//   mode 1: store f16, V  layout [b,h,d,s]   (transposed for the P@V B-operand)
//   mode 2: store fp32 flat [M,N]            (final output projection)
// Block = 256 threads (8 waves); tile 128x64; each wave computes 16x64.
// The shared 64x32 W-tile is double-buffered in LDS via async b128 copies
// (one global_load_async_to_lds_b128 per wave per K-step = 512B), so the
// K-step k+1 copy overlaps the K-step k WMMAs; synchronization is
// s_wait_asynccnt 0 + block barrier. Padded LDS rows (80B stride) make the
// B-fragment ds_load_b128 reads bank-conflict-free.
// ---------------------------------------------------------------------------
#define PRB 40  // padded B-tile row stride in halves (64B data + 16B pad)

__global__ __launch_bounds__(256) void gemm_wmma_kernel(
    const _Float16* __restrict__ A, const _Float16* __restrict__ W,
    const float* __restrict__ bias, _Float16* __restrict__ outh,
    float* __restrict__ outf, int mode) {
  __shared__ _Float16 bbuf[2][64 * PRB];

  const int tid  = threadIdx.x;
  const int lane = tid & 31;
  const int wave = tid >> 5;
  const int row  = lane & 15;
  const int hi   = lane >> 4;
  const int m0 = blockIdx.x * 128 + wave * 16;
  const int n0 = blockIdx.y * 64;

  // Async staging coordinates: thread t copies 16B of W row (n0 + t/4),
  // K-chunk (t%4)*8 halves. 256 threads cover the 64x32-half tile.
  const int br = tid >> 2;
  const int bc = (tid & 3) * 8;
  const _Float16* gsrc = W + (size_t)(n0 + br) * En + bc;
  const uint32_t ldst0 = lds_off(&bbuf[0][br * PRB + bc]);
  const uint32_t ldst1 = lds_off(&bbuf[1][br * PRB + bc]);

  // Prologue: stage k0=0 tile into buffer 0.
  async_copy_b128(ldst0, gsrc);
  wait_async0();
  __syncthreads();

  v8f acc0 = {}, acc1 = {}, acc2 = {}, acc3 = {};
  const _Float16* arow = A + (size_t)(m0 + row) * En;

  int cur = 0;
  for (int k0 = 0; k0 < En; k0 += 32) {
    if (k0 + 32 < En)  // prefetch next K-step into the other buffer
      async_copy_b128(cur ? ldst0 : ldst1, gsrc + k0 + 32);

    const _Float16* bb = bbuf[cur];
    v16h a  = load_a16(arow + k0 + hi * 16, arow + k0 + hi * 16 + 8);
    v16h b0 = *(const v16h*)(bb + (0 * 16 + row) * PRB + hi * 16);
    v16h b1 = *(const v16h*)(bb + (1 * 16 + row) * PRB + hi * 16);
    v16h b2 = *(const v16h*)(bb + (2 * 16 + row) * PRB + hi * 16);
    v16h b3 = *(const v16h*)(bb + (3 * 16 + row) * PRB + hi * 16);
    acc0 = wmma_f16(a, b0, acc0);
    acc1 = wmma_f16(a, b1, acc1);
    acc2 = wmma_f16(a, b2, acc2);
    acc3 = wmma_f16(a, b3, acc3);

    wait_async0();       // this wave's prefetch landed
    __syncthreads();     // everyone done reading cur + writing next
    cur ^= 1;
  }

  // C/D layout: reg r, lanes 0-15 -> (M=r, N=lane), lanes 16-31 -> (M=r+8).
#pragma unroll
  for (int nt = 0; nt < 4; ++nt) {
    const v8f acc = (nt == 0) ? acc0 : (nt == 1) ? acc1 : (nt == 2) ? acc2 : acc3;
    const int nn = n0 + nt * 16 + row;
    const float bv = bias[nn];
#pragma unroll
    for (int r = 0; r < 8; ++r) {
      const int mm = m0 + r + hi * 8;
      const float v = acc[r] + bv;
      if (mode == 2) {
        outf[(size_t)mm * En + nn] = v;
      } else {
        const int b_ = mm >> 11, s_ = mm & (Sn - 1);
        const int h_ = nn >> 6,  d_ = nn & (Dn - 1);
        const size_t dst = (mode == 0)
            ? (((size_t)(b_ * Hn + h_) * Sn + s_) * Dn + d_)   // [b,h,s,d]
            : (((size_t)(b_ * Hn + h_) * Dn + d_) * Sn + s_);  // [b,h,d,s]
        outh[dst] = (_Float16)v;
      }
    }
  }
}

// ---------------------------------------------------------------------------
// Causal flash attention. Grid: (S/64, B*H); block 128 (4 waves).
// Each wave owns 16 query rows; loops over 32-wide key blocks up to diagonal.
// Scores never hit memory; online softmax in fp32; P restaged via LDS to the
// WMMA A-operand layout (padded rows -> conflict-free ds_load_b128).
// ---------------------------------------------------------------------------
#define PROW 40  // 32 halves + 8 pad (row stride 80B -> conflict-free b128)

__global__ __launch_bounds__(128) void attn_kernel(
    const _Float16* __restrict__ Q, const _Float16* __restrict__ Km,
    const _Float16* __restrict__ Vt, _Float16* __restrict__ Ao) {
  __shared__ _Float16 pbuf[4][16 * PROW];

  const int lane = threadIdx.x & 31;
  const int wave = threadIdx.x >> 5;
  const int row  = lane & 15;
  const int hi   = lane >> 4;
  const int bh = blockIdx.y;
  const int q0 = blockIdx.x * 64;
  const int qw = q0 + wave * 16;

  const _Float16* Qp = Q  + (size_t)bh * Sn * Dn;  // [s,d]
  const _Float16* Kp = Km + (size_t)bh * Sn * Dn;  // [s,d]
  const _Float16* Vp = Vt + (size_t)bh * Dn * Sn;  // [d,s] (pre-transposed)
  _Float16* pb = pbuf[wave];

  // Q A-fragments for d = 0..31 and 32..63
  const _Float16* qrowp = Qp + (size_t)(qw + row) * Dn;
  const v16h qa0 = load_a16(qrowp + hi * 16,      qrowp + hi * 16 + 8);
  const v16h qa1 = load_a16(qrowp + 32 + hi * 16, qrowp + 32 + hi * 16 + 8);

  v8f o0 = {}, o1 = {}, o2 = {}, o3 = {};
  float mrow[8], lrow[8];
#pragma unroll
  for (int r = 0; r < 8; ++r) { mrow[r] = -3.0e38f; lrow[r] = 0.0f; }

  const int nkb = (q0 + 64) / 32;
  for (int jb = 0; jb < nkb; ++jb) {
    const int kb = jb * 32;

    // S = Q @ K^T for 16q x 32k (two 16-col subtiles, two K-steps of 32)
    v8f sL = {}, sR = {};
    {
      const _Float16* kl = Kp + (size_t)(kb + row) * Dn + hi * 16;
      const _Float16* kr = Kp + (size_t)(kb + 16 + row) * Dn + hi * 16;
      sL = wmma_f16(qa0, *(const v16h*)(kl),      sL);
      sL = wmma_f16(qa1, *(const v16h*)(kl + 32), sL);
      sR = wmma_f16(qa0, *(const v16h*)(kr),      sR);
      sR = wmma_f16(qa1, *(const v16h*)(kr + 32), sR);
    }

    // online softmax update (per output row; 16 lanes hold the 16 columns)
#pragma unroll
    for (int r = 0; r < 8; ++r) {
      const int qrow = qw + r + hi * 8;
      float sl = (kb + row      <= qrow) ? sL[r] * 0.125f : -3.0e38f;
      float sr = (kb + 16 + row <= qrow) ? sR[r] * 0.125f : -3.0e38f;
      float mx = fmaxf(sl, sr);
      mx = fmaxf(mx, __shfl_xor(mx, 1));
      mx = fmaxf(mx, __shfl_xor(mx, 2));
      mx = fmaxf(mx, __shfl_xor(mx, 4));
      mx = fmaxf(mx, __shfl_xor(mx, 8));
      const float mnew  = fmaxf(mrow[r], mx);
      const float alpha = __expf(mrow[r] - mnew);
      const float pl = __expf(sl - mnew);
      const float pr = __expf(sr - mnew);
      float rs = pl + pr;
      rs += __shfl_xor(rs, 1);
      rs += __shfl_xor(rs, 2);
      rs += __shfl_xor(rs, 4);
      rs += __shfl_xor(rs, 8);
      lrow[r] = lrow[r] * alpha + rs;
      mrow[r] = mnew;
      o0[r] *= alpha; o1[r] *= alpha; o2[r] *= alpha; o3[r] *= alpha;
      // stage P (C-frag layout -> row-major LDS tile)
      pb[(r + hi * 8) * PROW + row]      = (_Float16)pl;
      pb[(r + hi * 8) * PROW + 16 + row] = (_Float16)pr;
    }
    __syncthreads();

    // P as A-operand; V^T rows give contiguous B-operand columns.
    const v16h pa = load_a16(pb + row * PROW + hi * 16,
                             pb + row * PROW + hi * 16 + 8);
    {
      const _Float16* v0 = Vp + (size_t)(0 * 16 + row) * Sn + kb + hi * 16;
      const _Float16* v1 = Vp + (size_t)(1 * 16 + row) * Sn + kb + hi * 16;
      const _Float16* v2 = Vp + (size_t)(2 * 16 + row) * Sn + kb + hi * 16;
      const _Float16* v3 = Vp + (size_t)(3 * 16 + row) * Sn + kb + hi * 16;
      o0 = wmma_f16(pa, *(const v16h*)v0, o0);
      o1 = wmma_f16(pa, *(const v16h*)v1, o1);
      o2 = wmma_f16(pa, *(const v16h*)v2, o2);
      o3 = wmma_f16(pa, *(const v16h*)v3, o3);
    }
    __syncthreads();
  }

  // normalize and store f16 in [b,s,(h*D+d)] layout for the output projection
  const int b_ = bh / Hn, h_ = bh % Hn;
#pragma unroll
  for (int r = 0; r < 8; ++r) {
    const float inv = 1.0f / lrow[r];
    const int qrow = qw + r + hi * 8;
    _Float16* dst = Ao + ((size_t)(b_ * Sn + qrow)) * En + h_ * Dn;
    dst[0 * 16 + row] = (_Float16)(o0[r] * inv);
    dst[1 * 16 + row] = (_Float16)(o1[r] * inv);
    dst[2 * 16 + row] = (_Float16)(o2[r] * inv);
    dst[3 * 16 + row] = (_Float16)(o3[r] * inv);
  }
}

// ---------------------------------------------------------------------------
// Host-side launch. Workspace layout (f16 elems), total ~92 MB (fits in the
// 192 MB device L2, so inter-kernel staging stays on-chip):
//   xh[8M] | Wq/Wk/Wv/Wo[1M each] | Qh[8M] | Kh[8M] | Vt[8M] | Ao[8M]
// ---------------------------------------------------------------------------
extern "C" void kernel_launch(void* const* d_in, const int* in_sizes, int n_in,
                              void* d_out, int out_size, void* d_ws, size_t ws_size,
                              hipStream_t stream) {
  (void)in_sizes; (void)n_in; (void)out_size; (void)ws_size;
  const float* x  = (const float*)d_in[0];
  const float* Wq = (const float*)d_in[1];
  const float* bq = (const float*)d_in[2];
  const float* Wk = (const float*)d_in[3];
  const float* bk = (const float*)d_in[4];
  const float* Wv = (const float*)d_in[5];
  const float* bv = (const float*)d_in[6];
  const float* Wo = (const float*)d_in[7];
  const float* bo = (const float*)d_in[8];

  _Float16* xh  = (_Float16*)d_ws;
  _Float16* Wqh = xh  + (size_t)Mn * En;
  _Float16* Wkh = Wqh + (size_t)En * En;
  _Float16* Wvh = Wkh + (size_t)En * En;
  _Float16* Woh = Wvh + (size_t)En * En;
  _Float16* Qh  = Woh + (size_t)En * En;
  _Float16* Kh  = Qh  + (size_t)Mn * En;
  _Float16* Vth = Kh  + (size_t)Mn * En;
  _Float16* Aoh = Vth + (size_t)Mn * En;

  // fp32 -> f16
  cvt4_kernel<<<(Mn * En) / 1024, 256, 0, stream>>>(x,  xh,  (Mn * En) / 4);
  cvt4_kernel<<<(En * En) / 1024, 256, 0, stream>>>(Wq, Wqh, (En * En) / 4);
  cvt4_kernel<<<(En * En) / 1024, 256, 0, stream>>>(Wk, Wkh, (En * En) / 4);
  cvt4_kernel<<<(En * En) / 1024, 256, 0, stream>>>(Wv, Wvh, (En * En) / 4);
  cvt4_kernel<<<(En * En) / 1024, 256, 0, stream>>>(Wo, Woh, (En * En) / 4);

  // Q/K/V projections (V stored pre-transposed for P@V)
  dim3 g(Mn / 128, En / 64);
  gemm_wmma_kernel<<<g, 256, 0, stream>>>(xh, Wqh, bq, Qh,  nullptr, 0);
  gemm_wmma_kernel<<<g, 256, 0, stream>>>(xh, Wkh, bk, Kh,  nullptr, 0);
  gemm_wmma_kernel<<<g, 256, 0, stream>>>(xh, Wvh, bv, Vth, nullptr, 1);

  // causal flash attention
  attn_kernel<<<dim3(Sn / 64, Bn * Hn), 128, 0, stream>>>(Qh, Kh, Vth, Aoh);

  // output projection -> fp32
  gemm_wmma_kernel<<<g, 256, 0, stream>>>(Aoh, Woh, bo, nullptr, (float*)d_out, 2);
}
